// ClebschGordanTensorProductDense_11209864642657
// MI455X (gfx1250) — matrix operand order repaired and checked by reading, not compiled, not run on hardware
//
#include <hip/hip_runtime.h>
#include <math.h>

// ---------------------------------------------------------------------------
// e3nn Clebsch-Gordan tensor product, fused for MI455X (gfx1250, wave32).
// Stage 2 (channel matmul) uses v_wmma_f32_16x16x32_bf16; intermediate z
// lives only in LDS (bf16). Memory-bound: ~206MB HBM traffic -> ~9us ceiling.
// ---------------------------------------------------------------------------

typedef __attribute__((ext_vector_type(16))) __bf16 v16bf;
typedef __attribute__((ext_vector_type(8)))  float  v8f;

union Frag16 { v16bf v; uint4 q[2]; };

#define NROW 16          // n rows per block
#define NTHREADS 256     // 8 waves

// ------------------------- group / path metadata ---------------------------
constexpr int NG = 8;
// groups in sorted order: (0,+),(1,-),(1,+),(2,-),(2,+),(3,-),(3,+),(4,+)
constexpr int G_K3[NG]   = {1, 3, 3, 5, 5, 7, 7, 9};
constexpr int G_NP[NG]   = {3, 4, 2, 2, 4, 2, 1, 1};
constexpr int G_OUT[NG]  = {0, 128, 512, 896, 1536, 2176, 3072, 3968};
constexpr int G_FRAG[NG] = {0, 49152, 114688, 147456, 180224, 245760, 278528, 294912};
constexpr int WFRAG_TOTAL = 311296;          // sum U_g * 128
constexpr int OUT_ROW = 5120;                // output features per sample
constexpr int X1_ROW  = 1152;                // 9 * 128

// per-group path lists (bucket order == PATHS iteration order)
constexpr int PL1[NG][4] = {
  {0,1,2,0},{0,1,1,2},{1,2,0,0},{1,2,0,0},{0,1,2,2},{1,2,0,0},{2,0,0,0},{2,0,0,0}};
constexpr int PL2[NG][4] = {
  {0,1,2,0},{1,0,2,1},{1,2,0,0},{2,1,0,0},{2,1,0,2},{2,1,0,0},{2,0,0,0},{2,0,0,0}};
constexpr int PCG[NG][4] = {
  {0,44,600,0},{1,35,125,375},{53,625,0,0},{170,420,0,0},
  {10,80,350,700},{245,495,0,0},{825,0,0,0},{1000,0,0,0}};
constexpr int X1OFF[3] = {0, 128, 512};
constexpr int X2OFF[3] = {0, 1, 4};

constexpr int pbase(int g, int p) {
  int s = 0;
  for (int q = 0; q < p; ++q) s += (2 * PL1[g][q] + 1) * G_K3[g];
  return s;
}
constexpr int csz(int g) { return pbase(g, G_NP[g]); }
constexpr int smem_bytes(int g) {
  return NROW * 9 * 4 + NROW * csz(g) * 4 + G_K3[g] * NROW * (128 * G_NP[g]) * 2;
}

// ---------------------------------------------------------------------------
// Kernel 1: compute real Clebsch-Gordan tables on device (doubles), 1 thread
// per (l1,l2,l3) table. Mirrors the reference _su2_cg / _q / einsum exactly.
// ---------------------------------------------------------------------------
__device__ double dfact(int n) {
  double r = 1.0;
  for (int i = 2; i <= n; ++i) r *= (double)i;
  return r;
}

__device__ void su2_cg(int j1, int j2, int j3, double* C) {
  int d2 = 2 * j2 + 1, d3 = 2 * j3 + 1;
  int tot = (2 * j1 + 1) * d2 * d3;
  for (int t = 0; t < tot; ++t) C[t] = 0.0;
  if (!(abs(j1 - j2) <= j3 && j3 <= j1 + j2)) return;
  double pref0 = sqrt((double)(2 * j3 + 1) * dfact(j1 + j2 - j3) * dfact(j1 - j2 + j3)
                      * dfact(-j1 + j2 + j3) / dfact(j1 + j2 + j3 + 1));
  for (int m1 = -j1; m1 <= j1; ++m1)
    for (int m2 = -j2; m2 <= j2; ++m2) {
      int m3 = m1 + m2;
      if (abs(m3) > j3) continue;
      double pref = pref0 * sqrt(dfact(j3 + m3) * dfact(j3 - m3) * dfact(j1 - m1)
                                 * dfact(j1 + m1) * dfact(j2 - m2) * dfact(j2 + m2));
      double s = 0.0;
      for (int k = 0; k <= j1 + j2 - j3; ++k) {
        int a0 = k, a1 = j1 + j2 - j3 - k, a2 = j1 - m1 - k;
        int a3 = j2 + m2 - k, a4 = j3 - j2 + m1 + k, a5 = j3 - j1 - m2 + k;
        if (a0 < 0 || a1 < 0 || a2 < 0 || a3 < 0 || a4 < 0 || a5 < 0) continue;
        double denom = dfact(a0) * dfact(a1) * dfact(a2) * dfact(a3) * dfact(a4) * dfact(a5);
        s += ((k & 1) ? -1.0 : 1.0) / denom;
      }
      C[((j1 + m1) * d2 + (j2 + m2)) * d3 + (j3 + m3)] = pref * s;
    }
}

__device__ void qmat(int l, double* qr, double* qi) {
  int d = 2 * l + 1;
  for (int t = 0; t < d * d; ++t) { qr[t] = 0.0; qi[t] = 0.0; }
  double inv = 1.0 / sqrt(2.0);
  for (int m = -l; m < 0; ++m) {
    qr[(l + m) * d + (l - m)] = inv;   // q[l+m, l+|m|] = 1/sqrt2
    qi[(l + m) * d + (l + m)] = -inv;  // q[l+m, l-|m|] = -i/sqrt2
  }
  qr[l * d + l] = 1.0;
  for (int m = 1; m <= l; ++m) {
    double sgn = (m & 1) ? -1.0 : 1.0;
    qr[(l + m) * d + (l + m)] = sgn * inv;
    qi[(l + m) * d + (l - m)] = sgn * inv;
  }
  // multiply by (-i)^l
  double pr, pim;
  switch (l & 3) {
    case 0: pr = 1;  pim = 0;  break;
    case 1: pr = 0;  pim = -1; break;
    case 2: pr = -1; pim = 0;  break;
    default: pr = 0; pim = 1;  break;
  }
  for (int t = 0; t < d * d; ++t) {
    double r = qr[t], im = qi[t];
    qr[t] = r * pr - im * pim;
    qi[t] = r * pim + im * pr;
  }
}

__global__ void cg_kernel(float* __restrict__ cg) {
  int t = threadIdx.x;
  if (t >= 19) return;
  const int L1[19]  = {0,0,0,1,1,1,1,1,1,1,2,2,2,2,2,2,2,2,2};
  const int L2[19]  = {0,1,2,0,1,1,1,2,2,2,0,1,1,1,2,2,2,2,2};
  const int L3[19]  = {0,1,2,1,0,1,2,1,2,3,2,1,2,3,0,1,2,3,4};
  const int OFF[19] = {0,1,10,35,44,53,80,125,170,245,350,375,420,495,600,625,700,825,1000};
  int l1 = L1[t], l2 = L2[t], l3 = L3[t];
  int d1 = 2 * l1 + 1, d2 = 2 * l2 + 1, d3 = 2 * l3 + 1;
  double C[5 * 5 * 9];
  su2_cg(l1, l2, l3, C);
  double q1r[25], q1i[25], q2r[25], q2i[25], q3r[81], q3i[81];
  qmat(l1, q1r, q1i);
  qmat(l2, q2r, q2i);
  qmat(l3, q3r, q3i);
  // result[a][b][c] = Re sum_{i,k,n} q1[i][a] q2[k][b] conj(q3[n][c]) C[i][k][n]
  for (int a = 0; a < d1; ++a)
    for (int b = 0; b < d2; ++b)
      for (int c = 0; c < d3; ++c) {
        double re = 0.0;
        for (int i = 0; i < d1; ++i)
          for (int k = 0; k < d2; ++k)
            for (int n = 0; n < d3; ++n) {
              double cv = C[(i * d2 + k) * d3 + n];
              if (cv == 0.0) continue;
              double ar = q1r[i * d1 + a], ai = q1i[i * d1 + a];
              double br = q2r[k * d2 + b], bi = q2i[k * d2 + b];
              double cr = q3r[n * d3 + c], ci = -q3i[n * d3 + c];
              double xr = ar * br - ai * bi;
              double xi = ar * bi + ai * br;
              re += (xr * cr - xi * ci) * cv;
            }
        cg[OFF[t] + (a * d2 + b) * d3 + c] = (float)re;
      }
}

// ---------------------------------------------------------------------------
// Kernel 2: scale weights by 1/sqrt(U) and pre-swizzle to bf16 in WMMA
// B-fragment order used here, contiguous K, same as A (B 32x16: lanes 0-15 -> K 0-15, lanes 16-31 -> K 16-31,
// two K per VGPR in order). One thread per output element.
// ---------------------------------------------------------------------------
struct WPtrs { const float* p[8]; };

__global__ void wprep_kernel(WPtrs w, __bf16* __restrict__ wfrag) {
  int idx = blockIdx.x * NTHREADS + threadIdx.x;
  if (idx >= WFRAG_TOTAL) return;
  const int fragBase[NG + 1] = {0, 49152, 114688, 147456, 180224, 245760, 278528, 294912, 311296};
  const int np[NG] = {3, 4, 2, 2, 4, 2, 1, 1};
  int g = 0;
  while (g < NG - 1 && idx >= fragBase[g + 1]) ++g;
  int rel  = idx - fragBase[g];
  int e    = rel & 15;          // element within lane's fragment
  int lane = (rel >> 4) & 31;
  int f    = rel >> 9;          // fragment index = uc*8 + vt
  int uc   = f >> 3;
  int vt   = f & 7;
  int K    = ((lane >> 4) << 4) + e;   // lanes 0-15: K 0-15; lanes 16-31: K 16-31
  int u    = uc * 32 + K;
  int v    = vt * 16 + (lane & 15);
  int Ug   = 128 * np[g];
  float val = w.p[g][u * 128 + v] * rsqrtf((float)Ug);
  wfrag[idx] = (__bf16)val;
}

// ---------------------------------------------------------------------------
// Kernel 3 (per group G): fused CG contraction + bf16 WMMA channel matmul.
// Block = 16 samples x all 128 output channels. 8 waves, one v-tile each.
// ---------------------------------------------------------------------------
template <int G>
__global__ __launch_bounds__(NTHREADS) void tp_kernel(
    const float* __restrict__ x1, const float* __restrict__ x2,
    const float* __restrict__ cg, const __bf16* __restrict__ wfrag,
    float* __restrict__ out) {
  constexpr int K3 = G_K3[G];
  constexpr int NP = G_NP[G];
  constexpr int U = 128 * NP;
  constexpr int CSZ = csz(G);

  extern __shared__ char smem[];
  float*  sX2 = (float*)smem;                       // [16][9]
  float*  sC  = sX2 + NROW * 9;                     // [path][n][i][k] packed
  __bf16* sZ  = (__bf16*)(sC + NROW * CSZ);         // [k][n][u]

  const int tid = threadIdx.x;
  const int n0 = blockIdx.x * NROW;

  // ---- phase 0: stage x2 rows ----
  for (int t = tid; t < NROW * 9; t += NTHREADS) {
    int n = t / 9, j = t % 9;
    sX2[t] = x2[(size_t)(n0 + n) * 9 + j];
  }
  __syncthreads();

  // ---- phase 1: c_p[i,k] = sum_j x2[j] * cg[i,j,k] ----
#pragma unroll
  for (int p = 0; p < NP; ++p) {
    const int l1 = PL1[G][p], l2 = PL2[G][p];
    const int d1 = 2 * l1 + 1, d2 = 2 * l2 + 1;
    const int cgo = PCG[G][p], x2o = X2OFF[l2], pb = pbase(G, p);
    for (int t = tid; t < NROW * d1 * K3; t += NTHREADS) {
      int k = t % K3;
      int i = (t / K3) % d1;
      int n = t / (K3 * d1);
      float acc = 0.f;
      for (int j = 0; j < d2; ++j)
        acc += sX2[n * 9 + x2o + j] * cg[cgo + (i * d2 + j) * K3 + k];
      sC[pb * NROW + (n * d1 + i) * K3 + k] = acc;
    }
  }
  __syncthreads();

  // ---- phase 2: z[n,u,k] = sum_i x1[n,u,i] * c_p[i,k] -> LDS (bf16) ----
#pragma unroll
  for (int p = 0; p < NP; ++p) {
    const int l1 = PL1[G][p];
    const int d1 = 2 * l1 + 1;
    const int x1o = X1OFF[l1], pb = pbase(G, p);
    for (int t = tid; t < NROW * 128; t += NTHREADS) {
      int ul = t & 127;
      int n = t >> 7;
      const float* xr = x1 + (size_t)(n0 + n) * X1_ROW + x1o + ul * d1;
      float xv[5];
#pragma unroll
      for (int i = 0; i < d1; ++i) xv[i] = xr[i];
#pragma unroll
      for (int k = 0; k < K3; ++k) {
        float a = 0.f;
#pragma unroll
        for (int i = 0; i < d1; ++i)
          a += xv[i] * sC[pb * NROW + (n * d1 + i) * K3 + k];
        sZ[(k * NROW + n) * U + p * 128 + ul] = (__bf16)a;
      }
    }
  }
  __syncthreads();

  // ---- phase 3: y[n,v,k] = sum_u z * w  via v_wmma_f32_16x16x32_bf16 ----
  const int wave = tid >> 5;
  const int lane = tid & 31;
  const int vt = wave;              // 8 waves x 16 v-columns = 128
  const int hlf = lane >> 4;
  const int nv = lane & 15;

  v8f zero = {0.f, 0.f, 0.f, 0.f, 0.f, 0.f, 0.f, 0.f};
  v8f acc[K3];
#pragma unroll
  for (int k = 0; k < K3; ++k) acc[k] = zero;

  constexpr int NUC = U / 32;
  for (int uc = 0; uc < NUC; ++uc) {
    // B fragment: 2 coalesced b128 loads from pre-swizzled global weights
    Frag16 bfrag;
    const uint4* wp =
        (const uint4*)(wfrag + G_FRAG[G] + (size_t)(uc * 8 + vt) * 512 + lane * 16);
    bfrag.q[0] = wp[0];
    bfrag.q[1] = wp[1];
#pragma unroll
    for (int k = 0; k < K3; ++k) {
      // A fragment (16x32 bf16): lane holds row m=lane%16, K runs
      // [hlf*16, hlf*16+16) -> two ds_load_b128
      Frag16 afrag;
      const __bf16* za = sZ + (size_t)(k * NROW + nv) * U + uc * 32 + hlf * 16;
      afrag.q[0] = *(const uint4*)za;
      afrag.q[1] = *(const uint4*)(za + 8);
      acc[k] = __builtin_amdgcn_wmma_f32_16x16x32_bf16(
          false, afrag.v, false, bfrag.v, (short)0, acc[k], false, false);
    }
  }

  // ---- store: C layout VGPR r -> M = r + 8*hlf, lane%16 -> N ----
#pragma unroll
  for (int k = 0; k < K3; ++k) {
#pragma unroll
    for (int r = 0; r < 8; ++r) {
      int M = r + 8 * hlf;
      out[(size_t)(n0 + M) * OUT_ROW + G_OUT[G] + (vt * 16 + nv) * K3 + k] = acc[k][r];
    }
  }
}

// ---------------------------------------------------------------------------
extern "C" void kernel_launch(void* const* d_in, const int* in_sizes, int n_in,
                              void* d_out, int out_size, void* d_ws, size_t ws_size,
                              hipStream_t stream) {
  (void)in_sizes; (void)n_in; (void)out_size; (void)ws_size;
  const float* x1 = (const float*)d_in[0];
  const float* x2 = (const float*)d_in[1];
  WPtrs wp;
  for (int i = 0; i < 8; ++i) wp.p[i] = (const float*)d_in[2 + i];

  float*  cg    = (float*)d_ws;                       // 1225 floats
  __bf16* wfrag = (__bf16*)((char*)d_ws + 8192);      // 311296 bf16
  float*  out   = (float*)d_out;

  cg_kernel<<<1, 32, 0, stream>>>(cg);
  wprep_kernel<<<(WFRAG_TOTAL + NTHREADS - 1) / NTHREADS, NTHREADS, 0, stream>>>(wp, wfrag);

  constexpr int NBLK = 8192 / NROW;  // 512
  tp_kernel<0><<<NBLK, NTHREADS, smem_bytes(0), stream>>>(x1, x2, cg, wfrag, out);
  tp_kernel<1><<<NBLK, NTHREADS, smem_bytes(1), stream>>>(x1, x2, cg, wfrag, out);
  tp_kernel<2><<<NBLK, NTHREADS, smem_bytes(2), stream>>>(x1, x2, cg, wfrag, out);
  tp_kernel<3><<<NBLK, NTHREADS, smem_bytes(3), stream>>>(x1, x2, cg, wfrag, out);
  tp_kernel<4><<<NBLK, NTHREADS, smem_bytes(4), stream>>>(x1, x2, cg, wfrag, out);
  tp_kernel<5><<<NBLK, NTHREADS, smem_bytes(5), stream>>>(x1, x2, cg, wfrag, out);
  tp_kernel<6><<<NBLK, NTHREADS, smem_bytes(6), stream>>>(x1, x2, cg, wfrag, out);
  tp_kernel<7><<<NBLK, NTHREADS, smem_bytes(7), stream>>>(x1, x2, cg, wfrag, out);
}
